// MLPEncoder_8847632630416
// MI455X (gfx1250) — compile-verified, matrix operand order checked by reading, not run on hardware
//
#include <hip/hip_runtime.h>
#include <math.h>

#define N_NODES  50000
#define N_EDGES  800000
#define NE_TOT   (N_EDGES + N_NODES)     // self-loops appended
#define NC       (N_NODES * 64)
#define BN_EPS   1e-5f
#define NEG_SLOPE 0.2f

typedef __attribute__((ext_vector_type(16))) _Float16 v16h;
typedef __attribute__((ext_vector_type(8)))  float    v8f;

// ---------------------------------------------------------------------------
// helpers
// ---------------------------------------------------------------------------
__device__ __forceinline__ unsigned f2ord(float f) {
  unsigned u = __float_as_uint(f);
  return (u & 0x80000000u) ? ~u : (u | 0x80000000u);
}
__device__ __forceinline__ float ord2f(unsigned u) {
  unsigned b = (u & 0x80000000u) ? (u ^ 0x80000000u) : ~u;
  return __uint_as_float(b);
}
__device__ __forceinline__ void edge_sd(const int* __restrict__ ei, int e, int& s, int& d) {
  if (e < N_EDGES) { s = ei[e]; d = ei[N_EDGES + e]; }
  else             { s = d = e - N_EDGES; }          // self-loop
}
// K-index of fragment element idx for a lane whose K-half base is kb (0/8).
// ISA 16-bit A/B 16x32 layout: elements 0..7 -> K=kb+0..7, 8..15 -> K=16+kb+0..7
__device__ __forceinline__ int klocal(int idx, int kb) {
  return ((idx & 8) ? 16 : 0) + kb + (idx & 7);
}

// ---------------------------------------------------------------------------
// One-time weight pre-swizzle: W[K x 64] f32 -> per-lane WMMA B fragments f16.
// Layout: P[((ks*4 + nt)*32 + lane)*16 + i]  (one v16h per (ks,nt,lane))
// ---------------------------------------------------------------------------
__global__ void pack_w(const float* __restrict__ W, _Float16* __restrict__ P, int Ksteps) {
  int t = blockIdx.x * blockDim.x + threadIdx.x;
  if (t >= Ksteps * 4 * 32) return;
  int lane = t & 31;
  int nt   = (t >> 5) & 3;
  int ks   = t >> 7;
  int mrow = lane & 15;
  int kb   = (lane >> 4) << 3;
  _Float16* dst = P + (size_t)t * 16;
#pragma unroll
  for (int i = 0; i < 16; ++i) {
    int kk = ks * 32 + klocal(i, kb);
    dst[i] = (_Float16)W[kk * 64 + nt * 16 + mrow];
  }
}

// ---------------------------------------------------------------------------
// Dual GEMM via WMMA:  XL = X*Wl, XR = X*Wr
//   X : [50000 x K] fp32 row-major (K = 128 or 64); PWl/PWr: pre-swizzled f16
//   4 waves/block; one wave per 16-row tile; 4 col-tiles; K-step 32.
//   All 8 B fragments of a K-step are fetched as one batch so the compiler can
//   clause the 16 b128 loads and overlap the tail with WMMA issue.
// ---------------------------------------------------------------------------
__global__ void gemm_dual_wmma(const float* __restrict__ X,
                               const _Float16* __restrict__ PWl,
                               const _Float16* __restrict__ PWr,
                               float* __restrict__ XL,
                               float* __restrict__ XR,
                               int K) {
  const int lane = threadIdx.x & 31;
  const int tile = blockIdx.x * 4 + (threadIdx.x >> 5);
  if (tile >= 3125) return;                 // whole idle waves exit (EXEC stays all-1s)
  const int mrow = lane & 15;               // A: M row / B,C,D: N col within tile
  const int kb   = (lane >> 4) << 3;        // K-half base: 0 or 8
  v8f accL[4] = {};
  v8f accR[4] = {};
  const float* xrow = X + (size_t)(tile * 16 + mrow) * K;
  const v16h*  BL   = (const v16h*)PWl + lane;
  const v16h*  BR   = (const v16h*)PWr + lane;

  for (int ks = 0; ks * 32 < K; ++ks) {
    // A fragment: two contiguous 8-float runs per lane -> 4x b128 loads
    const float* p = xrow + ks * 32 + kb;
    float4 r0 = *(const float4*)(p);
    float4 r1 = *(const float4*)(p + 4);
    float4 r2 = *(const float4*)(p + 16);
    float4 r3 = *(const float4*)(p + 20);
    // batch all 8 B fragments of this K-step (independent -> one load clause)
    v16h bl[4], br[4];
#pragma unroll
    for (int nt = 0; nt < 4; ++nt) {
      bl[nt] = BL[(ks * 4 + nt) * 32];      // 32B aligned vector loads per lane
      br[nt] = BR[(ks * 4 + nt) * 32];
    }
    v16h a;
    a[0]  = (_Float16)r0.x; a[1]  = (_Float16)r0.y; a[2]  = (_Float16)r0.z; a[3]  = (_Float16)r0.w;
    a[4]  = (_Float16)r1.x; a[5]  = (_Float16)r1.y; a[6]  = (_Float16)r1.z; a[7]  = (_Float16)r1.w;
    a[8]  = (_Float16)r2.x; a[9]  = (_Float16)r2.y; a[10] = (_Float16)r2.z; a[11] = (_Float16)r2.w;
    a[12] = (_Float16)r3.x; a[13] = (_Float16)r3.y; a[14] = (_Float16)r3.z; a[15] = (_Float16)r3.w;
#pragma unroll
    for (int nt = 0; nt < 4; ++nt) {
      accL[nt] = __builtin_amdgcn_wmma_f32_16x16x32_f16(false, a, false, bl[nt],
                                                        (short)0, accL[nt], false, false);
      accR[nt] = __builtin_amdgcn_wmma_f32_16x16x32_f16(false, a, false, br[nt],
                                                        (short)0, accR[nt], false, false);
    }
  }
  // C/D layout: VGPR v, lanes 0-15 -> M=v, lanes 16-31 -> M=8+v ; N = lane&15
  const int orow = tile * 16 + ((lane >> 4) << 3);
#pragma unroll
  for (int nt = 0; nt < 4; ++nt) {
#pragma unroll
    for (int v = 0; v < 8; ++v) {
      const size_t o = (size_t)(orow + v) * 64 + nt * 16 + mrow;
      XL[o] = accL[nt][v];
      XR[o] = accR[nt][v];
    }
  }
}

// ---------------------------------------------------------------------------
// edge pass 1: e = att . leakyrelu(xl[src] + xr[dst]);  segment-max via u32 atomics
// ---------------------------------------------------------------------------
__global__ void edge_attn(const float* __restrict__ xl, const float* __restrict__ xr,
                          const int* __restrict__ ei, const float* __restrict__ att,
                          float* __restrict__ eb, unsigned* __restrict__ mb) {
  int e = blockIdx.x * blockDim.x + threadIdx.x;
  if (e >= NE_TOT) return;
  int s, d; edge_sd(ei, e, s, d);
  const float4* a = (const float4*)(xl + (size_t)s * 64);
  const float4* b = (const float4*)(xr + (size_t)d * 64);
  const float4* w = (const float4*)att;
  float acc = 0.f;
#pragma unroll
  for (int i = 0; i < 16; ++i) {
    float4 va = a[i], vb = b[i], vw = w[i];
    float h;
    h = va.x + vb.x; h = h > 0.f ? h : NEG_SLOPE * h; acc += h * vw.x;
    h = va.y + vb.y; h = h > 0.f ? h : NEG_SLOPE * h; acc += h * vw.y;
    h = va.z + vb.z; h = h > 0.f ? h : NEG_SLOPE * h; acc += h * vw.z;
    h = va.w + vb.w; h = h > 0.f ? h : NEG_SLOPE * h; acc += h * vw.w;
  }
  eb[e] = acc;
  atomicMax(mb + d, f2ord(acc));
}

__global__ void ord_to_float(unsigned* __restrict__ m) {
  int i = blockIdx.x * blockDim.x + threadIdx.x;
  if (i < N_NODES) {
    float f = ord2f(m[i]);
    ((float*)m)[i] = f;
  }
}

// edge pass 2: p = exp(e - m[dst]); segment-sum
__global__ void edge_softmax_p(float* __restrict__ eb, const float* __restrict__ mb,
                               float* __restrict__ sb, const int* __restrict__ ei) {
  int e = blockIdx.x * blockDim.x + threadIdx.x;
  if (e >= NE_TOT) return;
  int s, d; edge_sd(ei, e, s, d);
  float p = __expf(eb[e] - mb[d]);
  eb[e] = p;
  atomicAdd(sb + d, p);
}

// edge pass 3: out[dst] += (p/s[dst]) * xl[src]    (one wave per edge, lane=channel)
__global__ void edge_aggregate(const float* __restrict__ xl, const float* __restrict__ pb,
                               const float* __restrict__ sb, const int* __restrict__ ei,
                               float* __restrict__ outAgg) {
  int t = blockIdx.x * blockDim.x + threadIdx.x;
  int e = t >> 5, lane = t & 31;
  if (e >= NE_TOT) return;
  int s, d; edge_sd(ei, e, s, d);
  float alpha = pb[e] / (sb[d] + 1e-16f);
  const float* row = xl + (size_t)s * 64;
  float* orow = outAgg + (size_t)d * 64;
  atomicAdd(orow + lane,      alpha * row[lane]);
  atomicAdd(orow + lane + 32, alpha * row[lane + 32]);
}

// ---------------------------------------------------------------------------
// per-column sum / sum-of-squares (rows have fixed stride 64); LDS-atomic staged
// ---------------------------------------------------------------------------
__global__ void col_stats(const float* __restrict__ x, int ncols,
                          float* __restrict__ sum, float* __restrict__ sq) {
  __shared__ float ls[64], lq[64];
  if (threadIdx.x < 64) { ls[threadIdx.x] = 0.f; lq[threadIdx.x] = 0.f; }
  __syncthreads();
  int total = N_NODES * ncols;
  for (int i = blockIdx.x * blockDim.x + threadIdx.x; i < total;
       i += gridDim.x * blockDim.x) {
    int c = i % ncols, r = i / ncols;
    float v = x[(size_t)r * 64 + c];
    atomicAdd(&ls[c], v);
    atomicAdd(&lq[c], v * v);
  }
  __syncthreads();
  if (threadIdx.x < (unsigned)ncols) {
    atomicAdd(&sum[threadIdx.x], ls[threadIdx.x]);
    atomicAdd(&sq[threadIdx.x],  lq[threadIdx.x]);
  }
}

// BN (affine) + ReLU in place; st = [sum(64) | sumsq(64)]
__global__ void bn_relu_apply(float* __restrict__ h, const float* __restrict__ st,
                              const float* __restrict__ gamma, const float* __restrict__ beta) {
  const float invn = 1.f / (float)N_NODES;
  for (int i = blockIdx.x * blockDim.x + threadIdx.x; i < NC;
       i += gridDim.x * blockDim.x) {
    int c = i & 63;
    float mu  = st[c] * invn;
    float var = st[64 + c] * invn - mu * mu;
    float v = (h[i] - mu) * rsqrtf(var + BN_EPS) * gamma[c] + beta[c];
    h[i] = fmaxf(v, 0.f);
  }
}

// final head: mu = BN(out[:, :32], no affine);  sc = softplus(out[:, 32:] + b2[32:])
__global__ void final_head(const float* __restrict__ o2, const float* __restrict__ st,
                           const float* __restrict__ b2, float* __restrict__ out) {
  const float invn = 1.f / (float)N_NODES;
  const int total = N_NODES * 32;
  for (int i = blockIdx.x * blockDim.x + threadIdx.x; i < total;
       i += gridDim.x * blockDim.x) {
    int c = i & 31, r = i >> 5;
    float mu  = st[c] * invn;
    float var = st[64 + c] * invn - mu * mu;
    out[i] = (o2[(size_t)r * 64 + c] - mu) * rsqrtf(var + BN_EPS);
    float v = o2[(size_t)r * 64 + 32 + c] + b2[32 + c];
    out[total + i] = fmaxf(v, 0.f) + log1pf(expf(-fabsf(v)));   // stable softplus
  }
}

__global__ void zero_f32(float* __restrict__ p, int n) {
  for (int i = blockIdx.x * blockDim.x + threadIdx.x; i < n;
       i += gridDim.x * blockDim.x) p[i] = 0.f;
}

// ---------------------------------------------------------------------------
extern "C" void kernel_launch(void* const* d_in, const int* in_sizes, int n_in,
                              void* d_out, int out_size, void* d_ws, size_t ws_size,
                              hipStream_t stream) {
  const float* x      = (const float*)d_in[0];
  const int*   ei     = (const int*)  d_in[1];
  const float* Wl1    = (const float*)d_in[2];
  const float* Wr1    = (const float*)d_in[3];
  const float* att1   = (const float*)d_in[4];
  const float* gamma1 = (const float*)d_in[6];
  const float* beta1  = (const float*)d_in[7];
  const float* Wl2    = (const float*)d_in[8];
  const float* Wr2    = (const float*)d_in[9];
  const float* att2   = (const float*)d_in[10];
  const float* b2     = (const float*)d_in[11];
  float* out = (float*)d_out;

  // workspace carve-up: packed weights first (32B-aligned base), then f32 arrays
  _Float16* pwl1 = (_Float16*)d_ws;          // 4*4*32*16 = 8192 halves
  _Float16* pwr1 = pwl1 + 8192;              // 8192
  _Float16* pwl2 = pwr1 + 8192;              // 2*4*32*16 = 4096
  _Float16* pwr2 = pwl2 + 4096;              // 4096
  float* xl = (float*)(pwr2 + 4096);         // NC
  float* xr = xl + NC;                       // NC
  float* h  = xr + NC;                       // NC  (layer-1 aggregate, BN'd in place)
  float* o2 = h  + NC;                       // NC  (layer-2 aggregate)
  float* eb = o2 + NC;                       // NE_TOT (edge score, then p)
  float* mb = eb + NE_TOT;                   // N_NODES (segment max, u32 then f32)
  float* sb = mb + N_NODES;                  // N_NODES (segment sum)
  float* st = sb + N_NODES;                  // 128 (col sums | col sumsq)

  const int TB = 256;
  const int gEdge = (NE_TOT + TB - 1) / TB;
  const int gAgg  = (NE_TOT * 32 + TB - 1) / TB;
  const int gNode = (N_NODES + TB - 1) / TB;

  // one-time weight pre-swizzle into WMMA fragment order
  pack_w<<<2, TB, 0, stream>>>(Wl1, pwl1, 4);
  pack_w<<<2, TB, 0, stream>>>(Wr1, pwr1, 4);
  pack_w<<<1, TB, 0, stream>>>(Wl2, pwl2, 2);
  pack_w<<<1, TB, 0, stream>>>(Wr2, pwr2, 2);

  // init accumulators / stats (0.0f bit pattern == 0u, valid for ordered-u32 max)
  zero_f32<<<2048, TB, 0, stream>>>(h,  NC);
  zero_f32<<<2048, TB, 0, stream>>>(o2, NC);
  zero_f32<<<256,  TB, 0, stream>>>(mb, N_NODES);
  zero_f32<<<256,  TB, 0, stream>>>(sb, N_NODES);
  zero_f32<<<1,    TB, 0, stream>>>(st, 128);

  // ---- layer 1: GATv2(128 -> 64) ----
  gemm_dual_wmma<<<782, 128, 0, stream>>>(x, pwl1, pwr1, xl, xr, 128);
  edge_attn<<<gEdge, TB, 0, stream>>>(xl, xr, ei, att1, eb, (unsigned*)mb);
  ord_to_float<<<gNode, TB, 0, stream>>>((unsigned*)mb);
  edge_softmax_p<<<gEdge, TB, 0, stream>>>(eb, mb, sb, ei);
  edge_aggregate<<<gAgg, TB, 0, stream>>>(xl, eb, sb, ei, h);
  // b1 is absorbed by BN (mean subtracts it) -> skipped
  col_stats<<<1024, TB, 0, stream>>>(h, 64, st, st + 64);
  bn_relu_apply<<<2048, TB, 0, stream>>>(h, st, gamma1, beta1);

  // reset per-layer scratch
  zero_f32<<<256, TB, 0, stream>>>(mb, N_NODES);
  zero_f32<<<256, TB, 0, stream>>>(sb, N_NODES);
  zero_f32<<<1,   TB, 0, stream>>>(st, 128);

  // ---- layer 2: GATv2(64 -> 64), heads=1 so mean-over-heads is identity ----
  gemm_dual_wmma<<<782, 128, 0, stream>>>(h, pwl2, pwr2, xl, xr, 64);
  edge_attn<<<gEdge, TB, 0, stream>>>(xl, xr, ei, att2, eb, (unsigned*)mb);
  ord_to_float<<<gNode, TB, 0, stream>>>((unsigned*)mb);
  edge_softmax_p<<<gEdge, TB, 0, stream>>>(eb, mb, sb, ei);
  edge_aggregate<<<gAgg, TB, 0, stream>>>(xl, eb, sb, ei, o2);

  // ---- head: BN(no affine) on cols 0..31 (b2 cancels), softplus on cols 32..63 ----
  col_stats<<<1024, TB, 0, stream>>>(o2, 32, st, st + 64);
  final_head<<<2048, TB, 0, stream>>>(o2, st, b2, out);
}
